// SeparableAttentionPaper_20667382628681
// MI455X (gfx1250) — compile-verified, matrix operand order checked by reading, not run on hardware
//
#include <hip/hip_runtime.h>

#define DIM 1024
#define SEQ 4096
#define NB  8
#define ROWS (NB * SEQ)   // 32768

typedef __attribute__((ext_vector_type(16))) __bf16 v16bf;
typedef __attribute__((ext_vector_type(8)))  float  v8f;
typedef __attribute__((ext_vector_type(4)))  unsigned int u32x4;
typedef __attribute__((ext_vector_type(8)))  int i32x8;
typedef __attribute__((ext_vector_type(4)))  int i32x4;

__device__ __forceinline__ unsigned short f2bf(float f) {
    unsigned int u = __float_as_uint(f);
    u += 0x7FFFu + ((u >> 16) & 1u);          // round-to-nearest-even
    return (unsigned short)(u >> 16);
}
__device__ __forceinline__ float bf2f(unsigned short h) {
    return __uint_as_float(((unsigned int)h) << 16);
}

// ---------------------------------------------------------------------------
// Kernel 1: per-row dot with WI (-> s) and fp32 -> bf16 conversion of x.
// ---------------------------------------------------------------------------
__global__ __launch_bounds__(256)
void k_score_convert(const float* __restrict__ x, const float* __restrict__ WI,
                     unsigned short* __restrict__ xh, float* __restrict__ s) {
    __shared__ float red[256];
    const int row = blockIdx.x;
    const int t   = threadIdx.x;
    const float*    xr = x  + (size_t)row * DIM;
    unsigned short* xo = xh + (size_t)row * DIM;

    float4 v = *(const float4*)(xr + t * 4);
    float4 w = *(const float4*)(WI + t * 4);
    float acc = v.x * w.x + v.y * w.y + v.z * w.z + v.w * w.w;

    ushort4 h;
    h.x = f2bf(v.x); h.y = f2bf(v.y); h.z = f2bf(v.z); h.w = f2bf(v.w);
    *(ushort4*)(xo + t * 4) = h;

    red[t] = acc; __syncthreads();
    for (int o = 128; o > 0; o >>= 1) {
        if (t < o) red[t] += red[t + o];
        __syncthreads();
    }
    if (t == 0) s[row] = red[0];
}

// ---------------------------------------------------------------------------
// Kernel 2: elementwise fp32 -> bf16 (for WV, WO)
// ---------------------------------------------------------------------------
__global__ __launch_bounds__(256)
void k_f2bf(const float* __restrict__ src, unsigned short* __restrict__ dst, int n) {
    int i = blockIdx.x * 256 + threadIdx.x;
    if (i < n) dst[i] = f2bf(src[i]);
}

// ---------------------------------------------------------------------------
// Kernel 3: softmax over the sequence dim (4096) per batch. 1 block / batch.
// ---------------------------------------------------------------------------
__global__ __launch_bounds__(256)
void k_softmax(const float* __restrict__ s, float* __restrict__ cs) {
    __shared__ float red[256];
    const int b = blockIdx.x, t = threadIdx.x;
    const float* sb = s + (size_t)b * SEQ;

    float vals[16];
    float mx = -3.4e38f;
    #pragma unroll
    for (int i = 0; i < 16; ++i) { vals[i] = sb[t + 256 * i]; mx = fmaxf(mx, vals[i]); }
    red[t] = mx; __syncthreads();
    for (int o = 128; o > 0; o >>= 1) {
        if (t < o) red[t] = fmaxf(red[t], red[t + o]);
        __syncthreads();
    }
    mx = red[0]; __syncthreads();

    float sum = 0.f;
    #pragma unroll
    for (int i = 0; i < 16; ++i) { vals[i] = __expf(vals[i] - mx); sum += vals[i]; }
    red[t] = sum; __syncthreads();
    for (int o = 128; o > 0; o >>= 1) {
        if (t < o) red[t] += red[t + o];
        __syncthreads();
    }
    float inv = 1.f / red[0];
    #pragma unroll
    for (int i = 0; i < 16; ++i) cs[(size_t)b * SEQ + t + 256 * i] = vals[i] * inv;
}

// ---------------------------------------------------------------------------
// Kernel 4: u[b,k] = sum_n cs[b,n] * x[b,n,k]   (softmax folded into K-proj)
// ---------------------------------------------------------------------------
__global__ __launch_bounds__(256)
void k_weighted_sum(const unsigned short* __restrict__ xh,
                    const float* __restrict__ cs, float* __restrict__ u) {
    const int b = blockIdx.x >> 2;
    const int k = ((blockIdx.x & 3) << 8) + threadIdx.x;
    const unsigned short* xb = xh + (size_t)b * SEQ * DIM + k;
    const float*          cb = cs + (size_t)b * SEQ;
    float acc = 0.f;
    for (int n = 0; n < SEQ; ++n)
        acc += cb[n] * bf2f(xb[(size_t)n * DIM]);
    u[b * DIM + k] = acc;
}

// ---------------------------------------------------------------------------
// Kernel 5: cv[b,c] = sum_k u[b,k] * WK[c,k]   (tiny 8x1024x1024 GEMM, f32)
// ---------------------------------------------------------------------------
__global__ __launch_bounds__(256)
void k_cv(const float* __restrict__ u, const float* __restrict__ WK,
          float* __restrict__ cv) {
    const int tid = blockIdx.x * 256 + threadIdx.x;   // 0..8191
    const int b = tid >> 10, c = tid & 1023;
    const float* ub = u  + (size_t)b * DIM;
    const float* wr = WK + (size_t)c * DIM;
    float acc = 0.f;
    for (int k = 0; k < DIM; k += 4) {
        float4 uu = *(const float4*)(ub + k);
        float4 ww = *(const float4*)(wr + k);
        acc += uu.x * ww.x + uu.y * ww.y + uu.z * ww.z + uu.w * ww.w;
    }
    cv[tid] = acc;
}

// ---------------------------------------------------------------------------
// TDM: 2D tile DMA global->LDS. data_size=2B. Tile = tile_rows x 64 elements,
// row stride in memory = stride_elems. LDS padding: +16B after every 128B row
// (pad_interval code 4 = 32 DWORDs, pad_amount code 3 = 4 DWORDs) -> 144B row
// stride in LDS, which de-conflicts the 16B ds_load_b128 fragment reads.
// D# field packing per ISA 8.3/8.4 (group0: count/lds_addr/global_addr/type=2,
// group1: data_size, pads, dims, tile dims, dim0 stride).
// ---------------------------------------------------------------------------
#define KSTEP      64
#define LDS_ROW_B  144                 // 128B data + 16B pad
#define A_TILE_B   (128 * LDS_ROW_B)   // 18432
#define B_TILE_B   (64 * LDS_ROW_B)    //  9216
#define BUF_B      (A_TILE_B + B_TILE_B)

__device__ __forceinline__ void tdm_load_2d(unsigned lds_off,
                                            const unsigned short* gptr,
                                            int tile_rows, int stride_elems) {
    unsigned long long ga = (unsigned long long)(uintptr_t)gptr;
    u32x4 g0;
    g0[0] = 1u;                                       // count=1 (valid), user mode
    g0[1] = lds_off;                                  // lds_addr (bytes)
    g0[2] = (unsigned)(ga & 0xFFFFFFFFu);             // global_addr[31:0]
    g0[3] = (unsigned)((ga >> 32) & 0x01FFFFFFu)      // global_addr[56:32]
            | (2u << 30);                             // type=2 ("image")
    i32x8 g1;
    g1[0] = (int)((1u << 16)                          // data_size = 2 bytes
            | (1u << 20)                              // pad_enable
            | (4u << 22)                              // pad_interval: 32 DWORDs (128B)
            | (3u << 25));                            // pad_amount: 4 DWORDs (16B)
    g1[1] = (int)((unsigned)KSTEP << 16);             // tensor_dim0[15:0] = 64
    g1[2] = (int)((unsigned)(tile_rows & 0xFFFF) << 16); // tensor_dim1[15:0]
    g1[3] = (int)((unsigned)KSTEP << 16);             // tile_dim0 = 64
    g1[4] = tile_rows;                                // tile_dim1 (tile_dim2=0)
    g1[5] = stride_elems;                             // tensor_dim0_stride[31:0]
    g1[6] = 0;
    g1[7] = 0;
    i32x4 z4 = {0, 0, 0, 0};                          // groups 2/3 unused (2D)
#if defined(__clang_major__) && (__clang_major__ >= 23)
    i32x8 z8 = {0, 0, 0, 0, 0, 0, 0, 0};
    __builtin_amdgcn_tensor_load_to_lds(g0, g1, z4, z4, z8, 0);
#else
    __builtin_amdgcn_tensor_load_to_lds(g0, g1, z4, z4, 0);
#endif
}

// ---------------------------------------------------------------------------
// WMMA bf16 GEMM:  out[m,n] = sum_k A[m,k] * W[n,k]
//   MODE 0: epilogue  z = f2bf( cv[b,n] * relu(acc) )  -> bf16 zout
//   MODE 1: epilogue  y = acc                          -> f32 yout
// Block tile 128(M) x 64(N), 8 waves (4x2), wave tile 32x32 = 2x2 WMMA tiles.
// K staged in LDS via TDM, KSTEP=64, double buffered, TENSORcnt-synchronized.
// A-frag layout (ISA 7.12.2): lane<16 -> M=lane, K in {0..7,16..23};
// lane>=16 -> M=lane-16, K in {8..15,24..31}. B mirrors with N for M.
// ---------------------------------------------------------------------------
union Frag { v16bf v; uint4 q[2]; };

template <int MODE>
__global__ __launch_bounds__(256)
void k_gemm_bf16(const unsigned short* __restrict__ A,     // [ROWS, DIM] bf16
                 const unsigned short* __restrict__ W,     // [DIM, DIM]  bf16
                 const float* __restrict__ cv,             // [NB, DIM]   (MODE 0)
                 unsigned short* __restrict__ zout,        // bf16 out    (MODE 0)
                 float* __restrict__ yout) {               // f32 out     (MODE 1)
    __shared__ __align__(16) unsigned char lds[2 * BUF_B];

    const int lane = threadIdx.x & 31;
    const int wave = threadIdx.x >> 5;
    const int l16  = lane & 15;
    const int hi   = lane >> 4;        // 0 or 1
    const int wm   = wave & 3;         // 0..3  -> M
    const int wn   = wave >> 2;        // 0..1  -> N
    const int mB   = blockIdx.x * 128;
    const int nB   = blockIdx.y * 64;
    const int mW   = mB + wm * 32;
    const int nW   = nB + wn * 32;
    const unsigned lds_base = (unsigned)(uintptr_t)(&lds[0]);

    v8f acc[2][2];
    #pragma unroll
    for (int i = 0; i < 2; ++i)
        #pragma unroll
        for (int j = 0; j < 2; ++j)
            #pragma unroll
            for (int e = 0; e < 8; ++e) acc[i][j][e] = 0.f;

    const int NSTEPS = DIM / KSTEP;    // 16
    if (wave == 0) {                   // prime buffer 0
        tdm_load_2d(lds_base,            A + (size_t)mB * DIM, 128, DIM);
        tdm_load_2d(lds_base + A_TILE_B, W + (size_t)nB * DIM,  64, DIM);
    }

    for (int kk = 0; kk < NSTEPS; ++kk) {
        __syncthreads();               // everyone done reading buf[(kk+1)&1]
        if (wave == 0) {
            if (kk + 1 < NSTEPS) {
                const unsigned nb = lds_base + ((unsigned)((kk + 1) & 1)) * BUF_B;
                const int k0n = (kk + 1) * KSTEP;
                tdm_load_2d(nb,            A + (size_t)mB * DIM + k0n, 128, DIM);
                tdm_load_2d(nb + A_TILE_B, W + (size_t)nB * DIM + k0n,  64, DIM);
                __builtin_amdgcn_s_wait_tensorcnt(2);   // current step complete
            } else {
                __builtin_amdgcn_s_wait_tensorcnt(0);
            }
        }
        __syncthreads();               // buf[kk&1] visible to all waves

        const unsigned char* As = &lds[(unsigned)(kk & 1) * BUF_B];
        const unsigned char* Bs = As + A_TILE_B;

        #pragma unroll
        for (int kh = 0; kh < 2; ++kh) {          // two 32-wide K chunks
            Frag a[2], bb[2];
            #pragma unroll
            for (int mi = 0; mi < 2; ++mi) {
                const unsigned char* p =
                    As + (unsigned)(wm * 32 + mi * 16 + l16) * LDS_ROW_B
                       + (unsigned)(kh * 64 + hi * 16);
                a[mi].q[0] = *(const uint4*)p;          // K = base .. base+7
                a[mi].q[1] = *(const uint4*)(p + 32);   // K = base+16 .. base+23
            }
            #pragma unroll
            for (int ni = 0; ni < 2; ++ni) {
                const unsigned char* p =
                    Bs + (unsigned)(wn * 32 + ni * 16 + l16) * LDS_ROW_B
                       + (unsigned)(kh * 64 + hi * 16);
                bb[ni].q[0] = *(const uint4*)p;
                bb[ni].q[1] = *(const uint4*)(p + 32);
            }
            #pragma unroll
            for (int mi = 0; mi < 2; ++mi)
                #pragma unroll
                for (int ni = 0; ni < 2; ++ni)
                    acc[mi][ni] = __builtin_amdgcn_wmma_f32_16x16x32_bf16(
                        false, a[mi].v, false, bb[ni].v,
                        (short)0, acc[mi][ni], false, false);
        }
    }

    // C/D layout: VGPR j -> M = j (lanes 0-15) / j+8 (lanes 16-31); N = lane%16
    const int b = mB / SEQ;            // 4096 % 128 == 0: tile within one batch
    #pragma unroll
    for (int mi = 0; mi < 2; ++mi) {
        #pragma unroll
        for (int ni = 0; ni < 2; ++ni) {
            #pragma unroll
            for (int j = 0; j < 8; ++j) {
                const int m = mW + mi * 16 + hi * 8 + j;
                const int n = nW + ni * 16 + l16;
                float v = acc[mi][ni][j];
                if (MODE == 0) {
                    v = fmaxf(v, 0.0f) * cv[b * DIM + n];
                    zout[(size_t)m * DIM + n] = f2bf(v);
                } else {
                    yout[(size_t)m * DIM + n] = v;
                }
            }
        }
    }
}

// ---------------------------------------------------------------------------
// Launch
// ---------------------------------------------------------------------------
extern "C" void kernel_launch(void* const* d_in, const int* in_sizes, int n_in,
                              void* d_out, int out_size, void* d_ws, size_t ws_size,
                              hipStream_t stream) {
    const float* x  = (const float*)d_in[0];
    const float* WI = (const float*)d_in[1];
    const float* WK = (const float*)d_in[2];
    const float* WV = (const float*)d_in[3];
    const float* WO = (const float*)d_in[4];
    float* y = (float*)d_out;

    char* ws = (char*)d_ws;
    size_t off = 0;
    unsigned short* xh  = (unsigned short*)(ws + off); off += (size_t)ROWS * DIM * 2; // 64 MB
    unsigned short* zh  = (unsigned short*)(ws + off); off += (size_t)ROWS * DIM * 2; // 64 MB
    unsigned short* WVh = (unsigned short*)(ws + off); off += (size_t)DIM * DIM * 2;  //  2 MB
    unsigned short* WOh = (unsigned short*)(ws + off); off += (size_t)DIM * DIM * 2;  //  2 MB
    float* s   = (float*)(ws + off); off += (size_t)ROWS * 4;
    float* cs  = (float*)(ws + off); off += (size_t)ROWS * 4;
    float* u   = (float*)(ws + off); off += (size_t)NB * DIM * 4;
    float* cv  = (float*)(ws + off); off += (size_t)NB * DIM * 4;

    k_score_convert<<<ROWS, 256, 0, stream>>>(x, WI, xh, s);
    k_f2bf<<<(DIM * DIM) / 256, 256, 0, stream>>>(WV, WVh, DIM * DIM);
    k_f2bf<<<(DIM * DIM) / 256, 256, 0, stream>>>(WO, WOh, DIM * DIM);
    k_softmax<<<NB, 256, 0, stream>>>(s, cs);
    k_weighted_sum<<<NB * 4, 256, 0, stream>>>(xh, cs, u);
    k_cv<<<(NB * DIM) / 256, 256, 0, stream>>>(u, WK, cv);

    dim3 grid(ROWS / 128, DIM / 64);
    k_gemm_bf16<0><<<grid, 256, 0, stream>>>(xh, WVh, cv, zh, nullptr);
    k_gemm_bf16<1><<<grid, 256, 0, stream>>>(zh, WOh, nullptr, nullptr, y);
}
